// MultiHeadAttn_70325794504987
// MI455X (gfx1250) — matrix operand order repaired and checked by reading, not compiled, not run on hardware
//
#include <hip/hip_runtime.h>
#include <hip/hip_bf16.h>

// Problem constants
constexpr int kB  = 4;
constexpr int kS  = 2048;
constexpr int kD  = 1024;
constexpr int kH  = 16;
constexpr int kDH = 64;
constexpr int kTok = kB * kS;          // 8192 tokens
constexpr int kQKVN = 3 * kH * kDH;    // 3072

typedef __attribute__((ext_vector_type(16))) __bf16 bf16x16;
typedef __attribute__((ext_vector_type(8)))  float  f32x8;

// ---------- helpers ----------

__device__ inline unsigned short f2bf(float f) {
    unsigned u = __builtin_bit_cast(unsigned, f);
    return (unsigned short)((u + 0x7FFFu + ((u >> 16) & 1u)) >> 16);
}
__device__ inline unsigned pack2bf(float lo, float hi) {
    return (unsigned)f2bf(lo) | ((unsigned)f2bf(hi) << 16);
}
__device__ inline f32x8 zero8() {
    f32x8 z;
#pragma unroll
    for (int j = 0; j < 8; ++j) z[j] = 0.0f;
    return z;
}
__device__ inline f32x8 wmma_bf16(bf16x16 a, bf16x16 b, f32x8 c) {
    return __builtin_amdgcn_wmma_f32_16x16x32_bf16(
        /*neg_a=*/false, a, /*neg_b=*/false, b,
        /*c_mod=*/(short)0, c, /*reuse_a=*/false, /*reuse_b=*/false);
}

union Frag { bf16x16 v; unsigned u[8]; };

__device__ inline bf16x16 frag_from(uint4 a, uint4 b) {
    Frag f;
    f.u[0] = a.x; f.u[1] = a.y; f.u[2] = a.z; f.u[3] = a.w;
    f.u[4] = b.x; f.u[5] = b.y; f.u[6] = b.z; f.u[7] = b.w;
    return f.v;
}

// A fragment (16x32, MxK) from a [rows][32] bf16 tile (row stride 32 elems).
// VGPR i holds K = 2*i + 16*(lane>>4); row M = rowbase + (lane&15).
// -> one contiguous 32B run: u32 [hf*8 .. +7].
__device__ inline bf16x16 load_afrag(const unsigned short* tile, int rowbase, int lane) {
    int r   = rowbase + (lane & 15);
    int hf4 = ((lane >> 4) << 2);
    const unsigned* rowp = (const unsigned*)(tile + r * 32);
    uint4 lo = *(const uint4*)(rowp + 2 * hf4);
    uint4 hi = *(const uint4*)(rowp + 2 * hf4 + 4);
    return frag_from(lo, hi);
}

// B fragment (32x16, KxN) from a "Bt" tile stored [N rows][32 K] bf16.
// lanes 0-15: K=0..15 (VGPR i -> K=2i), lanes 16-31: K=16..31.
// -> one contiguous 32B run: u32 [hf*8 .. +7].
__device__ inline bf16x16 load_bfrag(const unsigned short* tile, int rowbase, int lane) {
    int n = rowbase + (lane & 15);
    const unsigned* rowp = (const unsigned*)(tile + n * 32) + ((lane >> 4) << 3);
    uint4 lo = *(const uint4*)rowp;
    uint4 hi = *(const uint4*)(rowp + 4);
    return frag_from(lo, hi);
}

// ---------- CDNA5 async global->LDS copy (16 bytes per lane) ----------
// VDST VGPR carries the LDS byte address (low 32 bits of the generic shared
// pointer, per the ISA's FLAT->LDS address truncation rule); GV mode 64-bit
// vaddr with saddr = off. Tracked by ASYNCcnt.
__device__ inline void async_copy16(const void* g, void* l) {
    unsigned loff = (unsigned)(unsigned long long)(uintptr_t)l;
    unsigned long long gaddr = (unsigned long long)(uintptr_t)g;
    asm volatile("global_load_async_to_lds_b128 %0, %1, off"
                 :: "v"(loff), "v"(gaddr) : "memory");
}
__device__ inline void async_wait0() {
    asm volatile("s_wait_asynccnt 0x0" ::: "memory");
}

// ---------- kernel 1: QKV projection ----------
// C[8192,3072] = inp[8192,1024] @ Wqkv^T + bqkv ; scatter bf16 into
// Q [B,H,S,DH], K [B,H,S,DH], Vt [B,H,DH,S].
__global__ __launch_bounds__(256) void qkv_kernel(
    const float* __restrict__ inp, const float* __restrict__ Wqkv,
    const float* __restrict__ bqkv,
    unsigned short* __restrict__ Q, unsigned short* __restrict__ K,
    unsigned short* __restrict__ Vt)
{
    __shared__ unsigned short Alds[128 * 32];
    __shared__ unsigned short Blds[128 * 32];

    const int tid  = threadIdx.x;
    const int lane = tid & 31;
    const int wave = tid >> 5;
    const int waveM = wave >> 2;   // 0..1  (64-row slice)
    const int waveN = wave & 3;    // 0..3  (32-col slice)
    const int m0 = blockIdx.y * 128;
    const int n0 = blockIdx.x * 128;

    f32x8 acc[4][2];
#pragma unroll
    for (int mi = 0; mi < 4; ++mi)
#pragma unroll
        for (int ni = 0; ni < 2; ++ni) acc[mi][ni] = zero8();

    for (int kb = 0; kb < kD; kb += 32) {
        __syncthreads();
        // cooperative load + fp32->bf16 convert: 128x32 tile = 2048 float2's
#pragma unroll
        for (int i = 0; i < 8; ++i) {
            int p   = tid + 256 * i;
            int row = p >> 4;
            int pc  = p & 15;
            float2 a = ((const float2*)(inp  + (size_t)(m0 + row) * kD + kb))[pc];
            float2 w = ((const float2*)(Wqkv + (size_t)(n0 + row) * kD + kb))[pc];
            ((unsigned*)Alds)[row * 16 + pc] = pack2bf(a.x, a.y);
            ((unsigned*)Blds)[row * 16 + pc] = pack2bf(w.x, w.y);
        }
        __syncthreads();

        bf16x16 af[4], bf[2];
#pragma unroll
        for (int mi = 0; mi < 4; ++mi) af[mi] = load_afrag(Alds, waveM * 64 + mi * 16, lane);
#pragma unroll
        for (int ni = 0; ni < 2; ++ni) bf[ni] = load_bfrag(Blds, waveN * 32 + ni * 16, lane);
#pragma unroll
        for (int mi = 0; mi < 4; ++mi)
#pragma unroll
            for (int ni = 0; ni < 2; ++ni)
                acc[mi][ni] = wmma_bf16(af[mi], bf[ni], acc[mi][ni]);
    }

    // epilogue: bias + scatter
    const int col = lane & 15;
    const int hf  = lane >> 4;
#pragma unroll
    for (int ni = 0; ni < 2; ++ni) {
        int n = n0 + waveN * 32 + ni * 16 + col;
        float bias = bqkv[n];
        int sec = n >> 10;          // 0=Q 1=K 2=V
        int wi  = n & 1023;
        int h   = wi >> 6;
        int d   = wi & 63;
#pragma unroll
        for (int mi = 0; mi < 4; ++mi) {
            int mbase = m0 + waveM * 64 + mi * 16 + hf * 8;
#pragma unroll
            for (int j = 0; j < 8; ++j) {
                int m = mbase + j;
                int b = m >> 11;
                int s = m & (kS - 1);
                unsigned short bv = f2bf(acc[mi][ni][j] + bias);
                size_t bh = (size_t)(b * kH + h);
                if (sec == 0)      Q [(bh * kS + s) * kDH + d] = bv;
                else if (sec == 1) K [(bh * kS + s) * kDH + d] = bv;
                else               Vt[(bh * kDH + d) * kS + s] = bv;
            }
        }
    }
}

// ---------- kernel 2: flash attention ----------
// one wave = 16 queries; block = 4 waves = 64 queries of one (b,h).
// K/V chunks are shared by all 4 waves -> staged once per block into LDS
// via async global->LDS DMA.
__global__ __launch_bounds__(128) void attn_kernel(
    const unsigned short* __restrict__ Q, const unsigned short* __restrict__ K,
    const unsigned short* __restrict__ Vt, const int* __restrict__ attn_mask,
    unsigned short* __restrict__ attnbuf)
{
    __shared__ unsigned short Klds[32 * 64];   // [key in chunk][d]   4 KB
    __shared__ unsigned short Vlds[64 * 32];   // [d][key in chunk]   4 KB
    __shared__ unsigned short Plds[4][16 * 32];

    const int tid  = threadIdx.x;
    const int lane = tid & 31;
    const int wave = tid >> 5;
    const int qb = blockIdx.x % (kS / 64);
    const int h  = (blockIdx.x / (kS / 64)) % kH;
    const int b  = blockIdx.x / ((kS / 64) * kH);
    const int qbase = qb * 64 + wave * 16;

    const unsigned short* Qp = Q  + (size_t)(b * kH + h) * kS * kDH;
    const unsigned short* Kp = K  + (size_t)(b * kH + h) * kS * kDH;
    const unsigned short* Vp = Vt + (size_t)(b * kH + h) * kDH * kS;

    const int r  = lane & 15;
    const int hf = lane >> 4;
    const float scale = 0.125f;  // 1/sqrt(64)
    const float NEG_INF = -__builtin_inff();

    // Q fragments for d=0..31 and d=32..63 (kept in registers)
    bf16x16 qf[2];
    {
        const unsigned* rowp = (const unsigned*)(Qp + (size_t)(qbase + r) * kDH);
        int hf4 = hf << 2;
        qf[0] = frag_from(*(const uint4*)(rowp + 2 * hf4),
                          *(const uint4*)(rowp + 4 + 2 * hf4));
        qf[1] = frag_from(*(const uint4*)(rowp + 16 + 2 * hf4),
                          *(const uint4*)(rowp + 20 + 2 * hf4));
    }

    float mrow[8], lrow[8];
    f32x8 o[4];
#pragma unroll
    for (int j = 0; j < 8; ++j) { mrow[j] = NEG_INF; lrow[j] = 0.0f; }
#pragma unroll
    for (int f = 0; f < 4; ++f) o[f] = zero8();

    for (int kc = 0; kc < kS; kc += 32) {
        __syncthreads();   // previous chunk fully consumed
        // stage K chunk (32x64 bf16) and V chunk (64x32 bf16): 2x 256 16B segs
#pragma unroll
        for (int i = 0; i < 2; ++i) {
            int p = tid + 128 * i;
            int krow = p >> 3, kseg = p & 7;
            async_copy16(Kp + (size_t)(kc + krow) * kDH + kseg * 8,
                         Klds + krow * 64 + kseg * 8);
            int vrow = p >> 2, vseg = p & 3;
            async_copy16(Vp + (size_t)vrow * kS + kc + vseg * 8,
                         Vlds + vrow * 32 + vseg * 8);
        }
        // key-padding mask (column index fixed per lane per score tile)
        const int mk0 = attn_mask[b * kS + kc + r];
        const int mk1 = attn_mask[b * kS + kc + 16 + r];
        async_wait0();
        __syncthreads();

        // scores: D = Q(16x64) @ K^T(64x32) in two 16-key tiles, K-loop of 2
        f32x8 s0 = zero8(), s1 = zero8();
#pragma unroll
        for (int t = 0; t < 2; ++t) {
            const unsigned* rowp = (const unsigned*)(Klds + (t * 16 + r) * 64)
                                   + (hf << 3);
            bf16x16 g0 = frag_from(*(const uint4*)rowp, *(const uint4*)(rowp + 4));
            bf16x16 g1 = frag_from(*(const uint4*)(rowp + 16), *(const uint4*)(rowp + 20));
            if (t == 0) { s0 = wmma_bf16(qf[0], g0, s0); s0 = wmma_bf16(qf[1], g1, s0); }
            else        { s1 = wmma_bf16(qf[0], g0, s1); s1 = wmma_bf16(qf[1], g1, s1); }
        }

        // online softmax over the 32-key chunk
        float p0[8], p1[8], rm[8];
#pragma unroll
        for (int j = 0; j < 8; ++j) {
            p0[j] = mk0 ? NEG_INF : s0[j] * scale;
            p1[j] = mk1 ? NEG_INF : s1[j] * scale;
            rm[j] = fmaxf(p0[j], p1[j]);
        }
#pragma unroll
        for (int off = 1; off < 16; off <<= 1)
#pragma unroll
            for (int j = 0; j < 8; ++j)
                rm[j] = fmaxf(rm[j], __shfl_xor(rm[j], off, 32));

        float alpha[8], rs[8];
#pragma unroll
        for (int j = 0; j < 8; ++j) {
            float mn = fmaxf(mrow[j], rm[j]);
            alpha[j] = (mrow[j] == NEG_INF) ? 0.0f : __expf(mrow[j] - mn);
            p0[j] = mk0 ? 0.0f : __expf(p0[j] - mn);
            p1[j] = mk1 ? 0.0f : __expf(p1[j] - mn);
            rs[j] = p0[j] + p1[j];
            mrow[j] = mn;
        }
#pragma unroll
        for (int off = 1; off < 16; off <<= 1)
#pragma unroll
            for (int j = 0; j < 8; ++j)
                rs[j] += __shfl_xor(rs[j], off, 32);
#pragma unroll
        for (int j = 0; j < 8; ++j) lrow[j] = lrow[j] * alpha[j] + rs[j];
#pragma unroll
        for (int f = 0; f < 4; ++f)
#pragma unroll
            for (int j = 0; j < 8; ++j) o[f][j] *= alpha[j];

        // re-layout P: C-layout -> A-layout via per-wave LDS (bf16 [16][32])
        unsigned short* pl = Plds[wave];
#pragma unroll
        for (int j = 0; j < 8; ++j) {
            int row = j + hf * 8;
            pl[row * 32 + r]      = f2bf(p0[j]);
            pl[row * 32 + 16 + r] = f2bf(p1[j]);
        }
        bf16x16 pf = load_afrag(pl, 0, lane);

        // O += P(16x32) @ V(32x64): Vlds rows are d, keys contiguous
#pragma unroll
        for (int f = 0; f < 4; ++f) {
            const unsigned* rowp = (const unsigned*)(Vlds + (f * 16 + r) * 32)
                                   + (hf << 3);
            bf16x16 vf = frag_from(*(const uint4*)rowp, *(const uint4*)(rowp + 4));
            o[f] = wmma_bf16(pf, vf, o[f]);
        }
    }

    // normalize and store attn output as bf16 [B,S,H*DH]
#pragma unroll
    for (int j = 0; j < 8; ++j) lrow[j] = 1.0f / lrow[j];
    size_t outbase = (size_t)b * kS * kD + (size_t)h * kDH;
#pragma unroll
    for (int f = 0; f < 4; ++f) {
        int d = f * 16 + r;
#pragma unroll
        for (int j = 0; j < 8; ++j) {
            int q = qbase + j + hf * 8;
            attnbuf[outbase + (size_t)q * kD + d] = f2bf(o[f][j] * lrow[j]);
        }
    }
}

// ---------- kernel 3: output projection ----------
// attn_out[8192,1024] = attnbuf[8192,1024](bf16) @ Wo^T   (f32 result)
// A-tile is already bf16 -> async global->LDS DMA, overlapped with the
// fp32->bf16 conversion of the Wo tile.
__global__ __launch_bounds__(256) void outproj_kernel(
    const unsigned short* __restrict__ attnbuf, const float* __restrict__ Wo,
    float* __restrict__ attn_out)
{
    __shared__ unsigned short Alds[128 * 32];
    __shared__ unsigned short Blds[128 * 32];

    const int tid  = threadIdx.x;
    const int lane = tid & 31;
    const int wave = tid >> 5;
    const int waveM = wave >> 2;
    const int waveN = wave & 3;
    const int m0 = blockIdx.y * 128;
    const int n0 = blockIdx.x * 128;

    f32x8 acc[4][2];
#pragma unroll
    for (int mi = 0; mi < 4; ++mi)
#pragma unroll
        for (int ni = 0; ni < 2; ++ni) acc[mi][ni] = zero8();

    for (int kb = 0; kb < kD; kb += 32) {
        __syncthreads();
        // A tile: 128x32 bf16 = 512 x 16B segments via async DMA
#pragma unroll
        for (int i = 0; i < 2; ++i) {
            int p   = tid + 256 * i;
            int row = p >> 2, seg = p & 3;
            async_copy16(attnbuf + (size_t)(m0 + row) * kD + kb + seg * 8,
                         Alds + row * 32 + seg * 8);
        }
        // B tile: fp32 -> bf16 convert in flight
#pragma unroll
        for (int i = 0; i < 8; ++i) {
            int p   = tid + 256 * i;
            int row = p >> 4;
            int pc  = p & 15;
            float2 w = ((const float2*)(Wo + (size_t)(n0 + row) * kD + kb))[pc];
            ((unsigned*)Blds)[row * 16 + pc] = pack2bf(w.x, w.y);
        }
        async_wait0();
        __syncthreads();

        bf16x16 af[4], bf[2];
#pragma unroll
        for (int mi = 0; mi < 4; ++mi) af[mi] = load_afrag(Alds, waveM * 64 + mi * 16, lane);
#pragma unroll
        for (int ni = 0; ni < 2; ++ni) bf[ni] = load_bfrag(Blds, waveN * 32 + ni * 16, lane);
#pragma unroll
        for (int mi = 0; mi < 4; ++mi)
#pragma unroll
            for (int ni = 0; ni < 2; ++ni)
                acc[mi][ni] = wmma_bf16(af[mi], bf[ni], acc[mi][ni]);
    }

    const int col = lane & 15;
    const int hf  = lane >> 4;
#pragma unroll
    for (int ni = 0; ni < 2; ++ni) {
        int n = n0 + waveN * 32 + ni * 16 + col;
#pragma unroll
        for (int mi = 0; mi < 4; ++mi) {
            int mbase = m0 + waveM * 64 + mi * 16 + hf * 8;
#pragma unroll
            for (int j = 0; j < 8; ++j)
                attn_out[(size_t)(mbase + j) * kD + n] = acc[mi][ni][j];
        }
    }
}

// ---------- kernel 4: residual + LayerNorm ----------
__global__ __launch_bounds__(256) void ln_kernel(
    const float* __restrict__ inp, const float* __restrict__ attn_out,
    const float* __restrict__ gamma, const float* __restrict__ beta,
    float* __restrict__ out)
{
    const int lane = threadIdx.x & 31;
    const int wave = threadIdx.x >> 5;
    const int tok  = blockIdx.x * 8 + wave;
    const float* xi = inp + (size_t)tok * kD;
    const float* xa = attn_out + (size_t)tok * kD;

    float x[32];
    float sum = 0.0f, ss = 0.0f;
#pragma unroll
    for (int i = 0; i < 32; ++i) {
        int c = lane + i * 32;
        float v = xi[c] + xa[c];
        x[i] = v; sum += v; ss += v * v;
    }
#pragma unroll
    for (int off = 1; off < 32; off <<= 1) {
        sum += __shfl_xor(sum, off, 32);
        ss  += __shfl_xor(ss,  off, 32);
    }
    float mu   = sum * (1.0f / kD);
    float var  = ss * (1.0f / kD) - mu * mu;
    float rstd = rsqrtf(var + 1e-5f);

    float* yo = out + (size_t)tok * kD;
#pragma unroll
    for (int i = 0; i < 32; ++i) {
        int c = lane + i * 32;
        yo[c] = (x[i] - mu) * rstd * gamma[c] + beta[c];
    }
}

// ---------- launch ----------
extern "C" void kernel_launch(void* const* d_in, const int* in_sizes, int n_in,
                              void* d_out, int out_size, void* d_ws, size_t ws_size,
                              hipStream_t stream)
{
    (void)in_sizes; (void)n_in; (void)out_size; (void)ws_size;
    const float* inp       = (const float*)d_in[0];
    const int*   attn_mask = (const int*)  d_in[1];
    const float* Wqkv      = (const float*)d_in[2];
    const float* bqkv      = (const float*)d_in[3];
    const float* Wo        = (const float*)d_in[4];
    const float* gamma     = (const float*)d_in[5];
    const float* beta      = (const float*)d_in[6];
    float* out = (float*)d_out;

    const size_t nElem = (size_t)kTok * kD;     // 8192*1024
    unsigned short* Qw      = (unsigned short*)d_ws;
    unsigned short* Kw      = Qw + nElem;
    unsigned short* Vtw     = Kw + nElem;
    unsigned short* attnbuf = Vtw + nElem;
    float* attn_out         = (float*)(attnbuf + nElem);

    qkv_kernel<<<dim3(kQKVN / 128, kTok / 128), 256, 0, stream>>>(
        inp, Wqkv, bqkv, Qw, Kw, Vtw);
    attn_kernel<<<dim3(kB * kH * (kS / 64)), 128, 0, stream>>>(
        Qw, Kw, Vtw, attn_mask, attnbuf);
    outproj_kernel<<<dim3(kD / 128, kTok / 128), 256, 0, stream>>>(
        attnbuf, Wo, attn_out);
    ln_kernel<<<dim3(kTok / 8), 256, 0, stream>>>(
        inp, attn_out, gamma, beta, out);
}
